// SetConv_86251533238887
// MI455X (gfx1250) — matrix operand order repaired and checked by reading, compiled, not run on hardware
//
#include <hip/hip_runtime.h>

// SetConv encoder for MI455X (gfx1250, wave32).
// out = concat(x_grid [B,128,128,2], z_grid [B,128,128,128]) as f32.
//
// Separable-weight trick: grid is a tensor product, so
//   W[(i,j), n] = Ex[b,i,n] * Ey[b,j,n]
// Kernel 1 builds Ex (f32) and Ey (f16) tables in workspace (2M exps instead
// of 134M). Kernel 2 does, per (b,i), the f16 WMMA GEMM
//   z_grid[b,i,j,:] = sum_n Ey[b,j,n] * (Ex[b,i,n] * z[b,n,:])
// with f32 accumulation via v_wmma_f32_16x16x32_f16.

typedef __attribute__((ext_vector_type(16))) _Float16 v16h;
typedef __attribute__((ext_vector_type(8)))  _Float16 v8h;
typedef __attribute__((ext_vector_type(4)))  _Float16 v4h;
typedef __attribute__((ext_vector_type(8)))  float    v8f;

#define B_   4
#define N_   2048
#define DZ_  128
#define G_   128
#define M_   (G_*G_)

#define TX_ELEMS (B_*G_*N_)         // 1,048,576 f32 (4 MB)
#define TY_OFF_BYTES (TX_ELEMS*4)   // f16 table after f32 table (2 MB)
// total workspace needed: 6 MB

// ---------------- Kernel 1: separable exp tables ----------------
__global__ __launch_bounds__(256)
void setconv_tables(const float* __restrict__ x,
                    const float* __restrict__ lsp,
                    float* __restrict__ tX,
                    _Float16* __restrict__ tY) {
  int t = blockIdx.x * blockDim.x + threadIdx.x;   // [0, 2*B*G*N)
  if (t >= 2 * TX_ELEMS) return;
  int d = (t >= TX_ELEMS) ? 1 : 0;   // 0 -> X table (dim0/i), 1 -> Y table (dim1/j)
  int e = d ? (t - TX_ELEMS) : t;    // flat [b][g][n]
  int n = e & (N_ - 1);
  int g = (e >> 11) & (G_ - 1);
  int b = e >> 18;

  float p   = lsp[d];
  float l   = 1e-5f + log1pf(__expf(p));   // 1e-5 + softplus
  float inv = 1.0f / (l * l);
  float axis = -2.0f + 4.0f * (float)g * (1.0f / 127.0f);  // linspace(-2,2,128)
  float xv   = x[(b * N_ + n) * 2 + d];
  float diff = axis - xv;
  float w    = __expf(-0.5f * diff * diff * inv);
  if (d == 0) tX[e] = w;
  else        tY[e] = (_Float16)w;
}

// ---------------- Kernel 2: per-(b,i) WMMA GEMM ----------------
#define LDS_STRIDE 40   // halfs per dz row: 32 k-values + 8 pad (conflict-free)

__global__ __launch_bounds__(256)
void setconv_gemm(const float* __restrict__ z,
                  const float* __restrict__ grid,
                  const float* __restrict__ tX,
                  const _Float16* __restrict__ tY,
                  float* __restrict__ out) {
  __shared__ _Float16 zl[DZ_ * LDS_STRIDE];   // 10,240 B

  int wg   = blockIdx.x;       // 0..511
  int b    = wg >> 7;
  int i    = wg & 127;
  int t    = threadIdx.x;
  int lane = t & 31;
  int wave = t >> 5;           // 0..7

  // x_grid: broadcast grid row i for this batch (256 floats per workgroup)
  out[b * (M_ * 2) + i * 256 + t] = grid[i * 256 + t];

  const float*    zb = z  + (size_t)b * N_ * DZ_;
  const float*    ex = tX + ((size_t)b * G_ + i) * N_;
  const _Float16* ty = tY + (size_t)b * G_ * N_;

  // staging role: this thread stages dz column sd, k-groups sg + {0,8,16,24}
  int sd = t & 127;
  int sg = (t >> 7) * 4;

  // WMMA fragment roles (ISA 7.12.2 16-bit layouts)
  int ln    = lane & 15;
  int hi    = lane >> 4;
  int jb    = wave * 16;
  const _Float16* arow = ty + (size_t)(jb + ln) * N_;
  int akoff = hi * 8;    // A: lanes 0-15 hold K 0..7/16..23; 16-31 hold 8..15/24..31
  int bkoff = hi * 8;    // B: lanes 0-15 hold K 0..7/16..23; 16-31 hold 8..15/24..31

  v8f acc[8];
#pragma unroll
  for (int q = 0; q < 8; ++q)
    acc[q] = (v8f){0.f, 0.f, 0.f, 0.f, 0.f, 0.f, 0.f, 0.f};

  for (int ks = 0; ks < N_ / 32; ++ks) {
    int k0 = ks * 32;
    __syncthreads();   // previous tile consumed

    // stage z' = Ex-scaled z, f32 -> f16, into LDS [dz][k] (stride 40)
#pragma unroll
    for (int q = 0; q < 4; ++q) {
      int kk = sg + q * 8;
      float e0 = ex[k0 + kk + 0], e1 = ex[k0 + kk + 1];
      float e2 = ex[k0 + kk + 2], e3 = ex[k0 + kk + 3];
      float z0 = zb[(size_t)(k0 + kk + 0) * DZ_ + sd];
      float z1 = zb[(size_t)(k0 + kk + 1) * DZ_ + sd];
      float z2 = zb[(size_t)(k0 + kk + 2) * DZ_ + sd];
      float z3 = zb[(size_t)(k0 + kk + 3) * DZ_ + sd];
      v4h pk;
      pk.x = (_Float16)(z0 * e0);
      pk.y = (_Float16)(z1 * e1);
      pk.z = (_Float16)(z2 * e2);
      pk.w = (_Float16)(z3 * e3);
      *(v4h*)&zl[sd * LDS_STRIDE + kk] = pk;   // ds_store_b64, conflict-free
    }
    __syncthreads();

    if (ks + 1 < N_ / 32) {   // hint next chunk into cache (global_prefetch_b8)
      __builtin_prefetch(&zb[(size_t)(k0 + 32 + sg) * DZ_ + sd], 0, 1);
      __builtin_prefetch(arow + k0 + 32 + akoff, 0, 1);
    }

    // A fragment: two 16B global loads from f16 Ey table
    union { v16h v; struct { v8h lo, hi; } p; } A;
    A.p.lo = *(const v8h*)(arow + k0 + akoff);
    A.p.hi = *(const v8h*)(arow + k0 + akoff + 16);

#pragma unroll
    for (int q = 0; q < 8; ++q) {
      int d0 = q * 16;
      union { v16h v; struct { v8h lo, hi; } p; } Bf;
      const _Float16* bptr = &zl[(size_t)(d0 + ln) * LDS_STRIDE + bkoff];
      Bf.p.lo = *(const v8h*)(bptr);       // ds_load_b128
      Bf.p.hi = *(const v8h*)(bptr + 16);  // ds_load_b128
      acc[q] = __builtin_amdgcn_wmma_f32_16x16x32_f16(
          false, A.v, false, Bf.v, (short)0, acc[q], false, false);
    }
  }

  // z_grid store: C/D layout — lane ln = N col, VGPR r = M row (+8 for hi lanes)
  float* oz = out + (size_t)B_ * M_ * 2 + ((size_t)(b * G_ + i) * G_) * DZ_;
#pragma unroll
  for (int q = 0; q < 8; ++q) {
    int d0 = q * 16;
#pragma unroll
    for (int r = 0; r < 8; ++r) {
      int jj = jb + hi * 8 + r;
      oz[(size_t)jj * DZ_ + d0 + ln] = acc[q][r];
    }
  }
}

// ---------------- launch ----------------
extern "C" void kernel_launch(void* const* d_in, const int* in_sizes, int n_in,
                              void* d_out, int out_size, void* d_ws, size_t ws_size,
                              hipStream_t stream) {
  const float* x    = (const float*)d_in[0];   // [4,2048,2]
  const float* z    = (const float*)d_in[1];   // [4,2048,128]
  const float* grid = (const float*)d_in[2];   // [128,128,2]
  const float* lsp  = (const float*)d_in[3];   // [2]
  float* out = (float*)d_out;

  float*    tX = (float*)d_ws;
  _Float16* tY = (_Float16*)((char*)d_ws + TY_OFF_BYTES);

  // Kernel 1: 2*B*G*N = 2,097,152 threads
  setconv_tables<<<(2 * TX_ELEMS) / 256, 256, 0, stream>>>(x, lsp, tX, tY);

  // Kernel 2: one workgroup per (b, i): 512 blocks x 256 threads
  setconv_gemm<<<B_ * G_, 256, 0, stream>>>(z, grid, tX, tY, out);
}
